// LinearTransformer_50285477101742
// MI455X (gfx1250) — compile-verified, matrix operand order checked by reading, not run on hardware
//
#include <hip/hip_runtime.h>

typedef __attribute__((ext_vector_type(16))) __bf16 v16bf;
typedef __attribute__((ext_vector_type(8)))  float  v8f;

#define BB      16
#define SS      4096
#define EE      128
#define DMODEL  64
#define NH      8
#define FFDIM   256
#define NLAYER  8
#define NCLS    1623
#define NT_OUT  102          // ceil(1623/16)
#define TOK     (BB*SS)

// ---------------- WMMA helpers ----------------

__device__ __forceinline__ v8f wmma_bf16(v16bf a, v16bf b, v8f c) {
  return __builtin_amdgcn_wmma_f32_16x16x32_bf16(false, a, false, b, (short)0, c, false, false);
}

// A fragment (16 rows x K=32 slice starting at k0) from a row-major bf16 tile in LDS.
// Layout (ISA 7.12.2, 16-bit A 16x32): lanes 0-15 hold M=lane, K={k0..k0+7, k0+16..k0+23};
// lanes 16-31 hold M=lane-16, K offset +8.
__device__ __forceinline__ v16bf load_a_frag(const __bf16* tile, int ld, int k0) {
  int lane = threadIdx.x & 31;
  int m    = lane & 15;
  int kb   = k0 + ((lane >> 4) << 3);
  const unsigned* row = (const unsigned*)(tile + m * ld);
  union { v16bf v; unsigned u[8]; } a;
  int h0 = kb >> 1, h1 = (kb + 16) >> 1;
#pragma unroll
  for (int i = 0; i < 4; ++i) { a.u[i] = row[h0 + i]; a.u[4 + i] = row[h1 + i]; }
  return a.v;
}

// B fragment from weights pre-packed in fragment order: one 32B contiguous load per lane.
__device__ __forceinline__ v16bf load_b_packed(const __bf16* packed, int tileIdx) {
  int lane = threadIdx.x & 31;
  return *(const v16bf*)(packed + ((size_t)tileIdx * 32 + lane) * 16);
}

__device__ __forceinline__ float phi_f(float x) { return x > 0.f ? x + 1.f : __expf(x); }
__device__ __forceinline__ float gelu_f(float x) {
  float x3 = x * x * x;
  return 0.5f * x * (1.f + tanhf(0.7978845608028654f * (x + 0.044715f * x3)));
}

// ---------------- weight packing ----------------
// Pack fp32 W (K x N row-major) into bf16 fragment-ordered tiles.
// tileIdx = nt*KT + kt ; per tile: 32 lanes x 16 elements, element e of lane:
//   n = nt*16 + (lane&15), k = kt*32 + ((lane>>4)*8) + (e<8 ? e : e+8)
__global__ void k_pack(const float* __restrict__ W, __bf16* __restrict__ out,
                       int K, int N, int KT, int NT) {
  int p = blockIdx.x * 256 + threadIdx.x;
  int total = KT * NT * 512;
  if (p >= total) return;
  int e = p & 15, lane = (p >> 4) & 31, t = p >> 9;
  int kt = t % KT, nt = t / KT;
  int n = nt * 16 + (lane & 15);
  int k = kt * 32 + ((lane >> 4) << 3) + ((e & 8) ? e + 8 : e);
  float v = (k < K && n < N) ? W[(size_t)k * N + n] : 0.f;
  out[p] = (__bf16)v;
}

// ---------------- embedding: (ex @ we + be + lemb[labels]) @ w_in + b_in ----------------
__global__ __launch_bounds__(128) void k_embed(
    const float* __restrict__ ex, const int* __restrict__ labels,
    const float* __restrict__ be, const float* __restrict__ lemb,
    const float* __restrict__ b_in,
    const __bf16* __restrict__ we_p, const __bf16* __restrict__ win_p,
    float* __restrict__ hh) {
  __shared__ __bf16 exs[64 * EE];
  __shared__ __bf16 mid[64 * EE];
  int t0 = blockIdx.x * 64;
  int tid = threadIdx.x;
  for (int i = tid; i < 64 * EE; i += 128) exs[i] = (__bf16)ex[(size_t)t0 * EE + i];
  __syncthreads();
  int wave = tid >> 5, lane = tid & 31;
  int col = lane & 15, rbase = (lane >> 4) << 3;
  const __bf16* exw = exs + wave * 16 * EE;
  v16bf a[4];
#pragma unroll
  for (int kt = 0; kt < 4; ++kt) a[kt] = load_a_frag(exw, EE, kt * 32);
#pragma unroll
  for (int nt = 0; nt < 8; ++nt) {
    v8f acc = {};
#pragma unroll
    for (int kt = 0; kt < 4; ++kt) acc = wmma_bf16(a[kt], load_b_packed(we_p, nt * 4 + kt), acc);
    int c = nt * 16 + col;
#pragma unroll
    for (int r = 0; r < 8; ++r) {
      int lr = wave * 16 + rbase + r;
      float v = acc[r] + be[c] + lemb[(size_t)labels[t0 + lr] * EE + c];
      mid[lr * EE + c] = (__bf16)v;
    }
  }
  __syncthreads();
  const __bf16* midw = mid + wave * 16 * EE;
  v16bf a2[4];
#pragma unroll
  for (int kt = 0; kt < 4; ++kt) a2[kt] = load_a_frag(midw, EE, kt * 32);
#pragma unroll
  for (int nt = 0; nt < 4; ++nt) {
    v8f acc = {};
#pragma unroll
    for (int kt = 0; kt < 4; ++kt) acc = wmma_bf16(a2[kt], load_b_packed(win_p, nt * 4 + kt), acc);
    int c = nt * 16 + col;
#pragma unroll
    for (int r = 0; r < 8; ++r) {
      int tok = t0 + wave * 16 + rbase + r;
      hh[(size_t)tok * DMODEL + c] = acc[r] + b_in[c];
    }
  }
}

// ---------------- per-layer phase A: mask, LN1, QKV, phi, kv/ksum reduction ----------------
__global__ __launch_bounds__(256) void k_qkv(
    const float* __restrict__ mask,
    const float* __restrict__ ln1s, const float* __restrict__ ln1b,
    const __bf16* __restrict__ wq_p, const float* __restrict__ bq,
    const __bf16* __restrict__ wk_p, const float* __restrict__ bk,
    const __bf16* __restrict__ wv_p, const float* __restrict__ bv,
    float* __restrict__ hh, float* __restrict__ pq,
    float* __restrict__ kv, float* __restrict__ ksum) {
  __shared__ __bf16 xs[128 * DMODEL];
  __shared__ __bf16 pks[128 * DMODEL];
  __shared__ __bf16 vs[128 * DMODEL];
  __shared__ float psum[256], psq[256], meanv[128], rstdv[128];
  int t0 = blockIdx.x * 128;
  int bidx = t0 >> 12;             // / S
  int tid = threadIdx.x;
  {
    int row = tid >> 1, half = tid & 1;
    int tok = t0 + row;
    float mval = mask[tok];
    float r[32]; float s = 0.f, q2 = 0.f;
    float* hp = hh + (size_t)tok * DMODEL + half * 32;
#pragma unroll
    for (int i = 0; i < 32; ++i) { float x = hp[i] * mval; r[i] = x; s += x; q2 += x * x; }
#pragma unroll
    for (int i = 0; i < 32; ++i) hp[i] = r[i];       // hh = hh * m3 (write back)
    psum[tid] = s; psq[tid] = q2;
    __syncthreads();
    if (half == 0) {
      float ts = psum[tid] + psum[tid + 1], tq = psq[tid] + psq[tid + 1];
      float m = ts * (1.f / 64.f);
      meanv[row] = m; rstdv[row] = rsqrtf(tq * (1.f / 64.f) - m * m + 1e-5f);
    }
    __syncthreads();
    float m = meanv[row], rs = rstdv[row];
#pragma unroll
    for (int i = 0; i < 32; ++i) {
      int c = half * 32 + i;
      xs[row * DMODEL + c] = (__bf16)((r[i] - m) * rs * ln1s[c] + ln1b[c]);
    }
  }
  __syncthreads();
  int wave = tid >> 5, lane = tid & 31;
  int col = lane & 15, rbase = (lane >> 4) << 3;
  const __bf16* xw = xs + wave * 16 * DMODEL;
  v16bf a0 = load_a_frag(xw, DMODEL, 0), a1 = load_a_frag(xw, DMODEL, 32);
#pragma unroll
  for (int nt = 0; nt < 4; ++nt) {            // Q -> phi -> pq (global fp32)
    v8f acc = {};
    acc = wmma_bf16(a0, load_b_packed(wq_p, nt * 2 + 0), acc);
    acc = wmma_bf16(a1, load_b_packed(wq_p, nt * 2 + 1), acc);
    int c = nt * 16 + col;
#pragma unroll
    for (int r = 0; r < 8; ++r) {
      int tok = t0 + wave * 16 + rbase + r;
      pq[(size_t)tok * DMODEL + c] = phi_f(acc[r] + bq[c]);
    }
  }
#pragma unroll
  for (int nt = 0; nt < 4; ++nt) {            // K -> phi*mask -> LDS
    v8f acc = {};
    acc = wmma_bf16(a0, load_b_packed(wk_p, nt * 2 + 0), acc);
    acc = wmma_bf16(a1, load_b_packed(wk_p, nt * 2 + 1), acc);
    int c = nt * 16 + col;
#pragma unroll
    for (int r = 0; r < 8; ++r) {
      int lr = wave * 16 + rbase + r;
      pks[lr * DMODEL + c] = (__bf16)(phi_f(acc[r] + bk[c]) * mask[t0 + lr]);
    }
  }
#pragma unroll
  for (int nt = 0; nt < 4; ++nt) {            // V -> LDS
    v8f acc = {};
    acc = wmma_bf16(a0, load_b_packed(wv_p, nt * 2 + 0), acc);
    acc = wmma_bf16(a1, load_b_packed(wv_p, nt * 2 + 1), acc);
    int c = nt * 16 + col;
#pragma unroll
    for (int r = 0; r < 8; ++r) {
      int lr = wave * 16 + rbase + r;
      vs[lr * DMODEL + c] = (__bf16)(acc[r] + bv[c]);
    }
  }
  __syncthreads();
  // kv[b,h,d,e] += sum_s pk*v   (512 entries over 256 threads)
#pragma unroll
  for (int rep = 0; rep < 2; ++rep) {
    int idx = tid + rep * 256;
    int h = idx >> 6, d = (idx >> 3) & 7, e = idx & 7;
    float s = 0.f;
    for (int srow = 0; srow < 128; ++srow)
      s += (float)pks[srow * DMODEL + h * 8 + d] * (float)vs[srow * DMODEL + h * 8 + e];
    atomicAdd(&kv[((size_t)bidx * NH + h) * 64 + d * 8 + e], s);
  }
  if (tid < 64) {
    int h = tid >> 3, d = tid & 7;
    float s = 0.f;
    for (int srow = 0; srow < 128; ++srow) s += (float)pks[srow * DMODEL + h * 8 + d];
    atomicAdd(&ksum[((size_t)bidx * NH + h) * 8 + d], s);
  }
}

// ---------------- per-layer phase B: attention out, wo, residual, LN2, FFN ----------------
__global__ __launch_bounds__(256) void k_attn_ffn(
    const float* __restrict__ pq,
    const float* __restrict__ kv, const float* __restrict__ ksum,
    const __bf16* __restrict__ wo_p, const float* __restrict__ bo,
    const float* __restrict__ ln2s, const float* __restrict__ ln2b,
    const __bf16* __restrict__ w1_p, const float* __restrict__ b1,
    const __bf16* __restrict__ w2_p, const float* __restrict__ b2,
    float* __restrict__ hh) {
  __shared__ __bf16 xbuf[128 * DMODEL];      // att, then y (LN2 output)
  __shared__ float  hh2[128 * DMODEL];       // post-attention residual stream
  __shared__ __bf16 gbuf[8 * 16 * 32];       // per-wave gelu staging (K-chunks of 32)
  __shared__ float  kvs[512], kss[64];
  __shared__ float  psum[256], psq[256], meanv[128], rstdv[128];
  int t0 = blockIdx.x * 128;
  int bidx = t0 >> 12;
  int tid = threadIdx.x;
  for (int i = tid; i < 512; i += 256) kvs[i] = kv[(size_t)bidx * NH * 64 + i];
  if (tid < 64) kss[tid] = ksum[(size_t)bidx * NH * 8 + tid];
  __syncthreads();
  {  // att = (pq . kv) / (pq . ksum + eps)
    int row = tid >> 1, half = tid & 1;
    const float* pr = pq + (size_t)(t0 + row) * DMODEL + half * 32;
    float pv[32];
#pragma unroll
    for (int i = 0; i < 32; ++i) pv[i] = pr[i];
#pragma unroll
    for (int hi = 0; hi < 4; ++hi) {
      int h = half * 4 + hi;
      float den = 1e-6f;
#pragma unroll
      for (int d = 0; d < 8; ++d) den += pv[hi * 8 + d] * kss[h * 8 + d];
      float inv = 1.f / den;
#pragma unroll
      for (int e = 0; e < 8; ++e) {
        float num = 0.f;
#pragma unroll
        for (int d = 0; d < 8; ++d) num += pv[hi * 8 + d] * kvs[(h * 8 + d) * 8 + e];
        xbuf[row * DMODEL + h * 8 + e] = (__bf16)(num * inv);
      }
    }
  }
  __syncthreads();
  int wave = tid >> 5, lane = tid & 31;
  int col = lane & 15, rbase = (lane >> 4) << 3;
  {  // hh2 = hh + att @ wo + bo
    const __bf16* xw = xbuf + wave * 16 * DMODEL;
    v16bf a0 = load_a_frag(xw, DMODEL, 0), a1 = load_a_frag(xw, DMODEL, 32);
#pragma unroll
    for (int nt = 0; nt < 4; ++nt) {
      v8f acc = {};
      acc = wmma_bf16(a0, load_b_packed(wo_p, nt * 2 + 0), acc);
      acc = wmma_bf16(a1, load_b_packed(wo_p, nt * 2 + 1), acc);
      int c = nt * 16 + col;
#pragma unroll
      for (int r = 0; r < 8; ++r) {
        int lr = wave * 16 + rbase + r;
        hh2[lr * DMODEL + c] = hh[(size_t)(t0 + lr) * DMODEL + c] + acc[r] + bo[c];
      }
    }
  }
  __syncthreads();
  {  // LN2 -> y in xbuf
    int row = tid >> 1, half = tid & 1;
    float r[32]; float s = 0.f, q2 = 0.f;
#pragma unroll
    for (int i = 0; i < 32; ++i) { float x = hh2[row * DMODEL + half * 32 + i]; r[i] = x; s += x; q2 += x * x; }
    psum[tid] = s; psq[tid] = q2;
    __syncthreads();
    if (half == 0) {
      float ts = psum[tid] + psum[tid + 1], tq = psq[tid] + psq[tid + 1];
      float m = ts * (1.f / 64.f);
      meanv[row] = m; rstdv[row] = rsqrtf(tq * (1.f / 64.f) - m * m + 1e-5f);
    }
    __syncthreads();
    float m = meanv[row], rs = rstdv[row];
#pragma unroll
    for (int i = 0; i < 32; ++i) {
      int c = half * 32 + i;
      xbuf[row * DMODEL + c] = (__bf16)((r[i] - m) * rs * ln2s[c] + ln2b[c]);
    }
  }
  __syncthreads();
  {  // FFN: hh = hh2 + gelu(y@w1+b1)@w2 + b2, streamed in K-chunks of 32
    const __bf16* yw = xbuf + wave * 16 * DMODEL;
    v16bf ay0 = load_a_frag(yw, DMODEL, 0), ay1 = load_a_frag(yw, DMODEL, 32);
    v8f outa[4];
#pragma unroll
    for (int nt = 0; nt < 4; ++nt) outa[nt] = (v8f){};
    __bf16* gw = gbuf + wave * 16 * 32;
    for (int j = 0; j < 8; ++j) {
#pragma unroll
      for (int tsub = 0; tsub < 2; ++tsub) {
        int nt1 = j * 2 + tsub;
        v8f g = {};
        g = wmma_bf16(ay0, load_b_packed(w1_p, nt1 * 2 + 0), g);
        g = wmma_bf16(ay1, load_b_packed(w1_p, nt1 * 2 + 1), g);
        int c = nt1 * 16 + col;
#pragma unroll
        for (int r = 0; r < 8; ++r)
          gw[(rbase + r) * 32 + tsub * 16 + col] = (__bf16)gelu_f(g[r] + b1[c]);
      }
      __syncthreads();
      v16bf ag = load_a_frag(gw, 32, 0);
#pragma unroll
      for (int nt = 0; nt < 4; ++nt)
        outa[nt] = wmma_bf16(ag, load_b_packed(w2_p, nt * 8 + j), outa[nt]);
      __syncthreads();
    }
#pragma unroll
    for (int nt = 0; nt < 4; ++nt) {
      int c = nt * 16 + col;
#pragma unroll
      for (int r = 0; r < 8; ++r) {
        int lr = wave * 16 + rbase + r;
        hh[(size_t)(t0 + lr) * DMODEL + c] = outa[nt][r] + b2[c] + hh2[lr * DMODEL + c];
      }
    }
  }
}

// ---------------- final: LN_f * mask, then @ w_out + b_out -> logits ----------------
__global__ __launch_bounds__(256) void k_final(
    const float* __restrict__ mask,
    const float* __restrict__ lnfs, const float* __restrict__ lnfb,
    const __bf16* __restrict__ wout_p, const float* __restrict__ b_out,
    const float* __restrict__ hh, float* __restrict__ out) {
  __shared__ __bf16 ys[128 * DMODEL];
  __shared__ float psum[256], psq[256], meanv[128], rstdv[128];
  int t0 = blockIdx.x * 128;
  int tid = threadIdx.x;
  {
    int row = tid >> 1, half = tid & 1;
    int tok = t0 + row;
    const float* hp = hh + (size_t)tok * DMODEL + half * 32;
    float r[32]; float s = 0.f, q2 = 0.f;
#pragma unroll
    for (int i = 0; i < 32; ++i) { float x = hp[i]; r[i] = x; s += x; q2 += x * x; }
    psum[tid] = s; psq[tid] = q2;
    __syncthreads();
    if (half == 0) {
      float ts = psum[tid] + psum[tid + 1], tq = psq[tid] + psq[tid + 1];
      float m = ts * (1.f / 64.f);
      meanv[row] = m; rstdv[row] = rsqrtf(tq * (1.f / 64.f) - m * m + 1e-5f);
    }
    __syncthreads();
    float m = meanv[row], rs = rstdv[row], mv = mask[tok];
#pragma unroll
    for (int i = 0; i < 32; ++i) {
      int c = half * 32 + i;
      ys[row * DMODEL + c] = (__bf16)(((r[i] - m) * rs * lnfs[c] + lnfb[c]) * mv);
    }
  }
  __syncthreads();
  int wave = tid >> 5, lane = tid & 31;
  int col = lane & 15, rbase = (lane >> 4) << 3;
  const __bf16* yw = ys + wave * 16 * DMODEL;
  v16bf a0 = load_a_frag(yw, DMODEL, 0), a1 = load_a_frag(yw, DMODEL, 32);
  for (int nt = 0; nt < NT_OUT; ++nt) {
    v8f acc = {};
    acc = wmma_bf16(a0, load_b_packed(wout_p, nt * 2 + 0), acc);
    acc = wmma_bf16(a1, load_b_packed(wout_p, nt * 2 + 1), acc);
    int c = nt * 16 + col;
    if (c < NCLS) {
      float bv = b_out[c];
#pragma unroll
      for (int r = 0; r < 8; ++r) {
        int tok = t0 + wave * 16 + rbase + r;
        out[(size_t)tok * NCLS + c] = acc[r] + bv;
      }
    }
  }
}

// ---------------- host launch ----------------
extern "C" void kernel_launch(void* const* d_in, const int* in_sizes, int n_in,
                              void* d_out, int out_size, void* d_ws, size_t ws_size,
                              hipStream_t stream) {
  (void)in_sizes; (void)n_in; (void)out_size; (void)ws_size;
  const float* ex    = (const float*)d_in[0];
  const int*   labels= (const int*)  d_in[1];
  const float* mask  = (const float*)d_in[2];
  const float* we    = (const float*)d_in[3];
  const float* be    = (const float*)d_in[4];
  const float* lemb  = (const float*)d_in[5];
  const float* w_in  = (const float*)d_in[6];
  const float* b_in  = (const float*)d_in[7];
  const float* ln1s  = (const float*)d_in[8];
  const float* ln1b  = (const float*)d_in[9];
  const float* wq    = (const float*)d_in[10];
  const float* bq    = (const float*)d_in[11];
  const float* wk    = (const float*)d_in[12];
  const float* bk    = (const float*)d_in[13];
  const float* wv    = (const float*)d_in[14];
  const float* bv    = (const float*)d_in[15];
  const float* wo    = (const float*)d_in[16];
  const float* bo    = (const float*)d_in[17];
  const float* ln2s  = (const float*)d_in[18];
  const float* ln2b  = (const float*)d_in[19];
  const float* w1    = (const float*)d_in[20];
  const float* b1    = (const float*)d_in[21];
  const float* w2    = (const float*)d_in[22];
  const float* b2    = (const float*)d_in[23];
  const float* lnfs  = (const float*)d_in[24];
  const float* lnfb  = (const float*)d_in[25];
  const float* wout  = (const float*)d_in[26];
  const float* b_out = (const float*)d_in[27];

  char* ws = (char*)d_ws;
  float* hh  = (float*)ws;                                   // 16 MB
  float* pq  = (float*)(ws + (size_t)16 * 1024 * 1024);      // 16 MB
  float* kvb = (float*)(ws + (size_t)32 * 1024 * 1024);      // L*B*H*64 f32
  float* ksb = kvb + (size_t)NLAYER * BB * NH * 64;          // L*B*H*8 f32
  __bf16* wp = (__bf16*)(ws + (size_t)33 * 1024 * 1024);
  __bf16* we_p  = wp;                 size_t o = 16384;
  __bf16* win_p = wp + o;             o += 8192;
  __bf16* wq_p  = wp + o;             o += (size_t)NLAYER * 4096;
  __bf16* wk_p  = wp + o;             o += (size_t)NLAYER * 4096;
  __bf16* wv_p  = wp + o;             o += (size_t)NLAYER * 4096;
  __bf16* wo_p  = wp + o;             o += (size_t)NLAYER * 4096;
  __bf16* w1_p  = wp + o;             o += (size_t)NLAYER * 16384;
  __bf16* w2_p  = wp + o;             o += (size_t)NLAYER * 16384;
  __bf16* wout_p= wp + o;

  auto pack = [&](const float* W, __bf16* dst, int K, int N, int KT, int NT) {
    int total = KT * NT * 512;
    k_pack<<<(total + 255) / 256, 256, 0, stream>>>(W, dst, K, N, KT, NT);
  };
  pack(we,   we_p,  128, 128, 4, 8);
  pack(w_in, win_p, 128, 64,  4, 4);
  for (int l = 0; l < NLAYER; ++l) {
    pack(wq + (size_t)l * 4096,  wq_p + (size_t)l * 4096,  64, 64, 2, 4);
    pack(wk + (size_t)l * 4096,  wk_p + (size_t)l * 4096,  64, 64, 2, 4);
    pack(wv + (size_t)l * 4096,  wv_p + (size_t)l * 4096,  64, 64, 2, 4);
    pack(wo + (size_t)l * 4096,  wo_p + (size_t)l * 4096,  64, 64, 2, 4);
    pack(w1 + (size_t)l * 16384, w1_p + (size_t)l * 16384, 64, 256, 2, 16);
    pack(w2 + (size_t)l * 16384, w2_p + (size_t)l * 16384, 256, 64, 8, 4);
  }
  pack(wout, wout_p, 64, NCLS, 2, NT_OUT);

  hipMemsetAsync(kvb, 0,
                 ((size_t)NLAYER * BB * NH * 64 + (size_t)NLAYER * BB * NH * 8) * 4,
                 stream);

  k_embed<<<TOK / 64, 128, 0, stream>>>(ex, labels, be, lemb, b_in, we_p, win_p, hh);

  for (int l = 0; l < NLAYER; ++l) {
    float* kvl = kvb + (size_t)l * BB * NH * 64;
    float* ksl = ksb + (size_t)l * BB * NH * 8;
    k_qkv<<<TOK / 128, 256, 0, stream>>>(
        mask, ln1s + l * DMODEL, ln1b + l * DMODEL,
        wq_p + (size_t)l * 4096, bq + l * DMODEL,
        wk_p + (size_t)l * 4096, bk + l * DMODEL,
        wv_p + (size_t)l * 4096, bv + l * DMODEL,
        hh, pq, kvl, ksl);
    k_attn_ffn<<<TOK / 128, 256, 0, stream>>>(
        pq, kvl, ksl,
        wo_p + (size_t)l * 4096, bo + l * DMODEL,
        ln2s + l * DMODEL, ln2b + l * DMODEL,
        w1_p + (size_t)l * 16384, b1 + l * FFDIM,
        w2_p + (size_t)l * 16384, b2 + l * DMODEL,
        hh);
  }

  k_final<<<TOK / 128, 256, 0, stream>>>(mask, lnfs, lnfb, wout_p, b_out, hh,
                                         (float*)d_out);
}
